// SwinTransformerBlock_76424648065607
// MI455X (gfx1250) — matrix operand order repaired and checked by reading, not compiled, not run on hardware
//
#include <hip/hip_runtime.h>
#include <stdint.h>

// ---------- types ----------
typedef __bf16 bf16_t;
typedef bf16_t   bf16x16 __attribute__((ext_vector_type(16)));
typedef float    f32x8   __attribute__((ext_vector_type(8)));
typedef uint32_t u32x4   __attribute__((ext_vector_type(4)));

#define WMMA_BF16(a, b, c) \
  __builtin_amdgcn_wmma_f32_16x16x32_bf16(false, (a), false, (b), (short)0, (c), false, false)

__device__ __forceinline__ uint16_t f2bf(float f) {
  union { float f; uint32_t u; } cv; cv.f = f;
  uint32_t u = cv.u;
  u += 0x7FFFu + ((u >> 16) & 1u);   // round-to-nearest-even
  return (uint16_t)(u >> 16);
}

union FragU { bf16x16 v; u32x4 q[2]; };

// A-matrix fragment (16xK tile, K-step 32): lane m=l%16, h=l/16
// a[i] = A[m][8h+i] (i<8),  A[m][16+8h+(i-8)] (i>=8)  -> two 16B loads
__device__ __forceinline__ bf16x16 load_frag_a(const uint16_t* base, int lda) {
  int lane = threadIdx.x & 31;
  int m = lane & 15, h = lane >> 4;
  FragU f;
  f.q[0] = *(const u32x4*)(base + (size_t)m * lda + 8 * h);
  f.q[1] = *(const u32x4*)(base + (size_t)m * lda + 16 + 8 * h);
  return f.v;
}
// B-matrix fragment from Bt (N rows, K contiguous): b[i] = Bt[n][8h+i] (i<8), Bt[n][16+8h+(i-8)] (i>=8)
__device__ __forceinline__ bf16x16 load_frag_b(const uint16_t* base, int ldb) {
  int lane = threadIdx.x & 31;
  int n = lane & 15, h = lane >> 4;
  FragU f;
  f.q[0] = *(const u32x4*)(base + (size_t)n * ldb + 8 * h);
  f.q[1] = *(const u32x4*)(base + (size_t)n * ldb + 16 + 8 * h);
  return f.v;
}

// shifted-window row (w*49+pos) -> original token index (incl. batch), ws=7 ss=3 H=W=56
__device__ __forceinline__ int win_row_to_token(int row) {
  int widx = row / 49, pos = row % 49;
  int b = widx >> 6, wi = widx & 63;
  int wh = wi >> 3, ww = wi & 7;
  int r = pos / 7, c = pos % 7;
  int ho = wh * 7 + r + 3; if (ho >= 56) ho -= 56;
  int wo = ww * 7 + c + 3; if (wo >= 56) wo -= 56;
  return b * 3136 + ho * 56 + wo;
}

// ---------- weight convert/transpose: (K,N) f32 -> (N,K) bf16 ----------
__global__ void wconv_kernel(const float* __restrict__ w, uint16_t* __restrict__ wT,
                             int K, int N) {
  int i = blockIdx.x * blockDim.x + threadIdx.x;
  if (i >= K * N) return;
  int k = i / N, n = i % N;
  wT[(size_t)n * K + k] = f2bf(w[i]);
}

// ---------- LayerNorm over C=96, wave-per-row, optional window gather ----------
__global__ void ln96_kernel(const float* __restrict__ x, const float* __restrict__ sc,
                            const float* __restrict__ bi, uint16_t* __restrict__ out,
                            int windowGather) {
  int wave = threadIdx.x >> 5;
  int lane = threadIdx.x & 31;
  int row = blockIdx.x * 8 + wave;
  if (row >= 12544) return;
  int src = windowGather ? win_row_to_token(row) : row;
  const float* p = x + (size_t)src * 96;
  float v0 = p[lane], v1 = p[lane + 32], v2 = p[lane + 64];
  float s = v0 + v1 + v2;
  for (int o = 16; o; o >>= 1) s += __shfl_xor(s, o, 32);
  float mu = s * (1.0f / 96.0f);
  float d0 = v0 - mu, d1 = v1 - mu, d2 = v2 - mu;
  float q = d0 * d0 + d1 * d1 + d2 * d2;
  for (int o = 16; o; o >>= 1) q += __shfl_xor(q, o, 32);
  float inv = rsqrtf(q * (1.0f / 96.0f) + 1e-6f);
  uint16_t* op = out + (size_t)row * 96;
  op[lane]      = f2bf(d0 * inv * sc[lane]      + bi[lane]);
  op[lane + 32] = f2bf(d1 * inv * sc[lane + 32] + bi[lane + 32]);
  op[lane + 64] = f2bf(d2 * inv * sc[lane + 64] + bi[lane + 64]);
}

// ---------- generic tiled bf16 GEMM: C = A(MxK) * Bt(NxK)^T + bias, fused epilogues ----
// EP 0: store bf16   1: GELU->bf16   2: f32 out = v + res   3: window-scatter f32 = v + res
template <int EP>
__global__ __launch_bounds__(128)
void gemm_kernel(const uint16_t* __restrict__ A, const uint16_t* __restrict__ Bt,
                 const float* __restrict__ bias, const float* __restrict__ res,
                 void* __restrict__ outp, int M, int N, int K) {
  __shared__ alignas(16) uint16_t sA[64 * 32];
  __shared__ alignas(16) uint16_t sB[32 * 32];
  int m0 = blockIdx.y * 64;
  int n0 = blockIdx.x * 32;
  int t = threadIdx.x;          // 128 threads = 4 waves
  int wv = t >> 5;
  f32x8 acc0 = {}, acc1 = {};
  for (int k0 = 0; k0 < K; k0 += 32) {
    __syncthreads();
    {
      int r0 = t >> 2, s0 = t & 3;               // A rows 0..31
      int r1 = (t + 128) >> 2;                   // A rows 32..63
      *(u32x4*)(sA + r0 * 32 + s0 * 8) =
          *(const u32x4*)(A + (size_t)(m0 + r0) * K + k0 + s0 * 8);
      *(u32x4*)(sA + r1 * 32 + s0 * 8) =
          *(const u32x4*)(A + (size_t)(m0 + r1) * K + k0 + s0 * 8);
      int rb = t >> 2, sb = t & 3;               // B rows 0..31
      *(u32x4*)(sB + rb * 32 + sb * 8) =
          *(const u32x4*)(Bt + (size_t)(n0 + rb) * K + k0 + sb * 8);
    }
    __syncthreads();
    bf16x16 a  = load_frag_a(sA + wv * 16 * 32, 32);
    bf16x16 b0 = load_frag_b(sB, 32);
    bf16x16 b1 = load_frag_b(sB + 16 * 32, 32);
    acc0 = WMMA_BF16(a, b0, acc0);
    acc1 = WMMA_BF16(a, b1, acc1);
  }
  int lane = t & 31, n = lane & 15, hh = lane >> 4;
  for (int nt = 0; nt < 2; ++nt) {
    f32x8 acc = nt ? acc1 : acc0;
    int col = n0 + nt * 16 + n;
    float bv = bias[col];
    for (int j = 0; j < 8; ++j) {
      int row = m0 + wv * 16 + hh * 8 + j;
      float v = acc[j] + bv;
      if (EP == 0) {
        ((uint16_t*)outp)[(size_t)row * N + col] = f2bf(v);
      } else if (EP == 1) {
        float g = 0.5f * v * (1.0f + tanhf(0.7978845608f * (v + 0.044715f * v * v * v)));
        ((uint16_t*)outp)[(size_t)row * N + col] = f2bf(g);
      } else if (EP == 2) {
        size_t idx = (size_t)row * N + col;
        ((float*)outp)[idx] = v + res[idx];
      } else {
        size_t idx = (size_t)win_row_to_token(row) * 96 + col;
        ((float*)outp)[idx] = v + res[idx];
      }
    }
  }
}

// ---------- window attention: one (window, head) per workgroup, N=49 pad 64, hd=32 -----
__global__ __launch_bounds__(128)
void winattn_kernel(const uint16_t* __restrict__ qkv, uint16_t* __restrict__ out) {
  __shared__ alignas(16) uint16_t sQ[64 * 32];
  __shared__ alignas(16) uint16_t sK[64 * 32];
  __shared__ alignas(16) uint16_t sVt[32 * 64];
  __shared__ alignas(16) float    sS[64 * 64];
  __shared__ alignas(16) uint16_t sP[64 * 64];
  __shared__ float sRs[64];
  __shared__ int   sCg[64];
  int hI = blockIdx.x;                 // head
  int widx = blockIdx.y;               // window (B*nW)
  int t = threadIdx.x, wv = t >> 5, lane = t & 31;
  int n = lane & 15, hh = lane >> 4;
  size_t rowbase = (size_t)widx * 49;
  int wi = widx & 63, wh = wi >> 3, ww = wi & 7;
  // column/row group ids for the shifted-window mask (-1 = padded)
  if (t < 64) {
    int g = -1;
    if (t < 49) {
      int r2 = t / 7, c2 = t % 7;
      g = ((wh < 7) ? 0 : (r2 < 4 ? 1 : 2)) * 3 + ((ww < 7) ? 0 : (c2 < 4 ? 1 : 2));
    }
    sCg[t] = g;
  }
  // stage Q, K (rows >=49 zero) and V transposed
  {
    int r = t >> 1, half = t & 1;
    u32x4 z = {};
    const uint16_t* qrow = qkv + (rowbase + r) * 288 + hI * 32 + half * 16;
    const uint16_t* krow = qrow + 96;
    *(u32x4*)(sQ + r * 32 + half * 16)     = (r < 49) ? *(const u32x4*)qrow : z;
    *(u32x4*)(sQ + r * 32 + half * 16 + 8) = (r < 49) ? *(const u32x4*)(qrow + 8) : z;
    *(u32x4*)(sK + r * 32 + half * 16)     = (r < 49) ? *(const u32x4*)krow : z;
    *(u32x4*)(sK + r * 32 + half * 16 + 8) = (r < 49) ? *(const u32x4*)(krow + 8) : z;
  }
  for (int e = t; e < 32 * 64; e += 128) {
    int d = e >> 6, kk = e & 63;
    sVt[d * 64 + kk] = (kk < 49) ? qkv[(rowbase + kk) * 288 + 192 + hI * 32 + d]
                                 : (uint16_t)0;
  }
  __syncthreads();
  // S = Q * K^T * scale
  bf16x16 aq = load_frag_a(sQ + wv * 16 * 32, 32);
  for (int nt = 0; nt < 4; ++nt) {
    f32x8 sacc = {};
    bf16x16 bk = load_frag_b(sK + nt * 16 * 32, 32);
    sacc = WMMA_BF16(aq, bk, sacc);
    for (int j = 0; j < 8; ++j)
      sS[(wv * 16 + hh * 8 + j) * 64 + nt * 16 + n] = sacc[j] * 0.17677669529663689f;
  }
  __syncthreads();
  // mask + softmax per row
  if (t < 64) {
    int row = t;
    if (row < 49) {
      int gi = sCg[row];
      float mx = -3e38f;
      for (int j2 = 0; j2 < 64; ++j2) {
        int gj = sCg[j2];
        float s = (gj < 0) ? -3e38f : sS[row * 64 + j2] + ((gi == gj) ? 0.0f : -100.0f);
        mx = fmaxf(mx, s);
      }
      float sum = 0.f;
      for (int j2 = 0; j2 < 64; ++j2) {
        int gj = sCg[j2];
        float s = (gj < 0) ? -3e38f : sS[row * 64 + j2] + ((gi == gj) ? 0.0f : -100.0f);
        float p = __expf(s - mx);
        sum += p;
        sP[row * 64 + j2] = f2bf(p);
      }
      sRs[row] = 1.0f / sum;
    } else {
      for (int j2 = 0; j2 < 64; ++j2) sP[row * 64 + j2] = 0;
      sRs[row] = 0.f;
    }
  }
  __syncthreads();
  // O = P * V
  f32x8 o0 = {}, o1 = {};
  for (int ks = 0; ks < 2; ++ks) {
    bf16x16 ap  = load_frag_a(sP + wv * 16 * 64 + ks * 32, 64);
    bf16x16 bv0 = load_frag_b(sVt + ks * 32, 64);
    bf16x16 bv1 = load_frag_b(sVt + 16 * 64 + ks * 32, 64);
    o0 = WMMA_BF16(ap, bv0, o0);
    o1 = WMMA_BF16(ap, bv1, o1);
  }
  for (int j = 0; j < 8; ++j) {
    int row = wv * 16 + hh * 8 + j;
    if (row < 49) {
      float rs = sRs[row];
      size_t orow = (rowbase + row) * 96 + hI * 32;
      out[orow + n]      = f2bf(o0[j] * rs);
      out[orow + 16 + n] = f2bf(o1[j] * rs);
    }
  }
}

// ---------- global attention, flash-style: (b, head, 64-query tile) per workgroup ------
__global__ __launch_bounds__(128)
void flashattn_kernel(const uint16_t* __restrict__ qkv, uint16_t* __restrict__ out) {
  __shared__ alignas(16) uint16_t sK[64 * 32];
  __shared__ alignas(16) uint16_t sVt[32 * 64];
  __shared__ alignas(16) float    sS[64 * 64];
  __shared__ alignas(16) uint16_t sP[64 * 64];
  __shared__ float sM[64], sL[64], sAl[64];
  const int L = 3136;
  int q0 = blockIdx.x * 64;
  int b = blockIdx.y / 3, hI = blockIdx.y % 3;
  int t = threadIdx.x, wv = t >> 5, lane = t & 31;
  int n = lane & 15, hh = lane >> 4;
  size_t base = (size_t)b * L;
  // Q fragment stays in registers for the whole KV stream (K dim = hd = 32)
  bf16x16 aq = load_frag_a(qkv + (base + q0 + wv * 16) * 288 + hI * 32, 288);
  f32x8 o0 = {}, o1 = {};
  if (t < 64) { sM[t] = -3e38f; sL[t] = 0.f; }
  for (int kt = 0; kt < 49; ++kt) {
    size_t krow0 = base + (size_t)kt * 64;
    __syncthreads();
    {
      int r = t >> 1, half = t & 1;
      const uint16_t* src = qkv + (krow0 + r) * 288 + 96 + hI * 32 + half * 16;
      *(u32x4*)(sK + r * 32 + half * 16)     = *(const u32x4*)src;
      *(u32x4*)(sK + r * 32 + half * 16 + 8) = *(const u32x4*)(src + 8);
    }
    for (int e = t; e < 32 * 64; e += 128) {
      int d = e >> 6, kk = e & 63;
      sVt[d * 64 + kk] = qkv[(krow0 + kk) * 288 + 192 + hI * 32 + d];
    }
    __syncthreads();
    for (int nt = 0; nt < 4; ++nt) {
      f32x8 sacc = {};
      bf16x16 bk = load_frag_b(sK + nt * 16 * 32, 32);
      sacc = WMMA_BF16(aq, bk, sacc);
      for (int j = 0; j < 8; ++j)
        sS[(wv * 16 + hh * 8 + j) * 64 + nt * 16 + n] = sacc[j] * 0.17677669529663689f;
    }
    __syncthreads();
    if (t < 64) {              // online softmax stats
      int row = t;
      float m_old = sM[row];
      float mx = m_old;
      for (int j2 = 0; j2 < 64; ++j2) mx = fmaxf(mx, sS[row * 64 + j2]);
      float alpha = __expf(m_old - mx);
      float psum = 0.f;
      for (int j2 = 0; j2 < 64; ++j2) {
        float p = __expf(sS[row * 64 + j2] - mx);
        psum += p;
        sP[row * 64 + j2] = f2bf(p);
      }
      sM[row] = mx;
      sL[row] = sL[row] * alpha + psum;
      sAl[row] = alpha;
    }
    __syncthreads();
    for (int j = 0; j < 8; ++j) {   // rescale running O
      float a = sAl[wv * 16 + hh * 8 + j];
      o0[j] *= a; o1[j] *= a;
    }
    for (int ks = 0; ks < 2; ++ks) {
      bf16x16 ap  = load_frag_a(sP + wv * 16 * 64 + ks * 32, 64);
      bf16x16 bv0 = load_frag_b(sVt + ks * 32, 64);
      bf16x16 bv1 = load_frag_b(sVt + 16 * 64 + ks * 32, 64);
      o0 = WMMA_BF16(ap, bv0, o0);
      o1 = WMMA_BF16(ap, bv1, o1);
    }
  }
  __syncthreads();
  for (int j = 0; j < 8; ++j) {
    int row = wv * 16 + hh * 8 + j;
    float inv = 1.0f / sL[row];
    size_t orow = (base + q0 + row) * 96 + hI * 32;
    out[orow + n]      = f2bf(o0[j] * inv);
    out[orow + 16 + n] = f2bf(o1[j] * inv);
  }
}

__global__ void tail_kernel(float* out, int off) {
  if (threadIdx.x == 0) { out[off] = 56.0f; out[off + 1] = 56.0f; }
}

// ---------- host orchestration ----------
extern "C" void kernel_launch(void* const* d_in, const int* in_sizes, int n_in,
                              void* d_out, int out_size, void* d_ws, size_t ws_size,
                              hipStream_t stream) {
  const float* x      = (const float*)d_in[0];
  const float* n1s    = (const float*)d_in[3];
  const float* n1b    = (const float*)d_in[4];
  const float* qkv_w  = (const float*)d_in[5];
  const float* qkv_b  = (const float*)d_in[6];
  const float* proj_w = (const float*)d_in[7];
  const float* proj_b = (const float*)d_in[8];
  const float* l1s    = (const float*)d_in[9];
  const float* l1b    = (const float*)d_in[10];
  const float* qkv2_w = (const float*)d_in[11];
  const float* qkv2_b = (const float*)d_in[12];
  const float* out_w  = (const float*)d_in[13];
  const float* out_b  = (const float*)d_in[14];
  const float* l2s    = (const float*)d_in[15];
  const float* l2b    = (const float*)d_in[16];
  const float* fc1_w  = (const float*)d_in[17];
  const float* fc1_b  = (const float*)d_in[18];
  const float* fc2_w  = (const float*)d_in[19];
  const float* fc2_b  = (const float*)d_in[20];

  const int M = 12544;
  char* wsp = (char*)d_ws;
  size_t off = 0;
  auto alloc = [&](size_t bytes) -> void* {
    void* p = wsp + off;
    off = (off + bytes + 255) & ~(size_t)255;
    return p;
  };
  uint16_t* wT_qkv  = (uint16_t*)alloc((size_t)288 * 96 * 2);
  uint16_t* wT_proj = (uint16_t*)alloc((size_t)96 * 96 * 2);
  uint16_t* wT_qkv2 = (uint16_t*)alloc((size_t)288 * 96 * 2);
  uint16_t* wT_out  = (uint16_t*)alloc((size_t)96 * 96 * 2);
  uint16_t* wT_fc1  = (uint16_t*)alloc((size_t)384 * 96 * 2);
  uint16_t* wT_fc2  = (uint16_t*)alloc((size_t)96 * 384 * 2);
  uint16_t* a_ln    = (uint16_t*)alloc((size_t)M * 96 * 2);
  uint16_t* qkvb    = (uint16_t*)alloc((size_t)M * 288 * 2);
  uint16_t* attn_o  = (uint16_t*)alloc((size_t)M * 96 * 2);
  float*    x1      = (float*)   alloc((size_t)M * 96 * 4);
  uint16_t* y_ln    = (uint16_t*)alloc((size_t)M * 96 * 2);
  uint16_t* qkv2b   = (uint16_t*)alloc((size_t)M * 288 * 2);
  uint16_t* attn2_o = (uint16_t*)alloc((size_t)M * 96 * 2);
  float*    x2      = (float*)   alloc((size_t)M * 96 * 4);
  uint16_t* z_ln    = (uint16_t*)alloc((size_t)M * 96 * 2);
  uint16_t* h1      = (uint16_t*)alloc((size_t)M * 384 * 2);
  (void)n_in; (void)in_sizes; (void)ws_size;

  wconv_kernel<<<(96 * 288 + 255) / 256, 256, 0, stream>>>(qkv_w, wT_qkv, 96, 288);
  wconv_kernel<<<(96 * 96 + 255) / 256, 256, 0, stream>>>(proj_w, wT_proj, 96, 96);
  wconv_kernel<<<(96 * 288 + 255) / 256, 256, 0, stream>>>(qkv2_w, wT_qkv2, 96, 288);
  wconv_kernel<<<(96 * 96 + 255) / 256, 256, 0, stream>>>(out_w, wT_out, 96, 96);
  wconv_kernel<<<(96 * 384 + 255) / 256, 256, 0, stream>>>(fc1_w, wT_fc1, 96, 384);
  wconv_kernel<<<(384 * 96 + 255) / 256, 256, 0, stream>>>(fc2_w, wT_fc2, 384, 96);

  // LN1 fused with shift+window partition
  ln96_kernel<<<M / 8, 256, 0, stream>>>(x, n1s, n1b, a_ln, 1);
  // window QKV
  gemm_kernel<0><<<dim3(288 / 32, M / 64), 128, 0, stream>>>(a_ln, wT_qkv, qkv_b, nullptr, qkvb, M, 288, 96);
  // masked window attention
  winattn_kernel<<<dim3(3, 256), 128, 0, stream>>>(qkvb, attn_o);
  // proj + window-reverse scatter + residual (x1 = x + out)
  gemm_kernel<3><<<dim3(96 / 32, M / 64), 128, 0, stream>>>(attn_o, wT_proj, proj_b, x, x1, M, 96, 96);
  // trailing block: LN, QKV
  ln96_kernel<<<M / 8, 256, 0, stream>>>(x1, l1s, l1b, y_ln, 0);
  gemm_kernel<0><<<dim3(288 / 32, M / 64), 128, 0, stream>>>(y_ln, wT_qkv2, qkv2_b, nullptr, qkv2b, M, 288, 96);
  // flash global attention over L=3136 per (b, head)
  flashattn_kernel<<<dim3(49, 12), 128, 0, stream>>>(qkv2b, attn2_o);
  // out proj + residual (x2 = x1 + ao@W + b)
  gemm_kernel<2><<<dim3(96 / 32, M / 64), 128, 0, stream>>>(attn2_o, wT_out, out_b, x1, x2, M, 96, 96);
  // MLP
  ln96_kernel<<<M / 8, 256, 0, stream>>>(x2, l2s, l2b, z_ln, 0);
  gemm_kernel<1><<<dim3(384 / 32, M / 64), 128, 0, stream>>>(z_ln, wT_fc1, fc1_b, nullptr, h1, M, 384, 96);
  gemm_kernel<2><<<dim3(96 / 32, M / 64), 128, 0, stream>>>(h1, wT_fc2, fc2_b, x2, (float*)d_out, M, 96, 384);
  // tuple tail: (H, W) = (56, 56)
  if (out_size >= 2) tail_kernel<<<1, 32, 0, stream>>>((float*)d_out, out_size - 2);
}
